// MultiHeadAttention_7687991460016
// MI455X (gfx1250) — matrix operand order repaired and checked by reading, not compiled, not run on hardware
//
#include <hip/hip_runtime.h>
#include <hip/hip_bf16.h>

// ---------------------------------------------------------------------------
// MultiHeadAttention forward for MI455X (gfx1250), bf16 WMMA pipeline.
//   1) convert q + 4 weight matrices f32 -> bf16
//   2) Q = (q Wq^T) * (0.125*log2e)   [bf16, row major]
//      K = (q Wk^T)                    [bf16, row major]
//      Vt= (q Wv^T) transposed/head    [bf16, [B][H][64][L]]
//   3) flash attention per (b,h,16-row wave) using v_wmma_f32_16x16x32_bf16
//   4) out = attn Wo^T                 [f32 to d_out]
// Workspace: 48 MB.
// ---------------------------------------------------------------------------

constexpr int kDIM  = 1024;
constexpr int kNH   = 16;
constexpr int kHD   = 64;
constexpr int kL    = 2048;
constexpr int kB    = 2;
constexpr int kTOK  = kB * kL;          // 4096 rows in all GEMMs

typedef __bf16 bf16;
typedef __attribute__((ext_vector_type(4)))  float f32x4;
typedef __attribute__((ext_vector_type(8)))  float f32x8;
typedef __attribute__((ext_vector_type(4)))  bf16  bf16x4;
typedef __attribute__((ext_vector_type(8)))  bf16  bf16x8;
typedef __attribute__((ext_vector_type(16))) bf16  bf16x16;

static __device__ __forceinline__ bf16x16 join16(bf16x8 lo, bf16x8 hi) {
  return __builtin_shufflevector(lo, hi, 0,1,2,3,4,5,6,7,8,9,10,11,12,13,14,15);
}

static __device__ __forceinline__ f32x8 wmma_bf16(bf16x16 a, bf16x16 b, f32x8 c) {
  return __builtin_amdgcn_wmma_f32_16x16x32_bf16(
      /*neg_a=*/false, a, /*neg_b=*/false, b,
      /*c_mod=*/(short)0, c, /*reuse_a=*/false, /*reuse_b=*/false);
}

// lanes (i, i+16) exchange: combine softmax stats held by the two half-groups.
static __device__ __forceinline__ float swz_xor16(float x) {
  // group-of-32 swizzle: and=0x1f, or=0, xor=0x10 -> SWAPX16
  return __builtin_bit_cast(float,
      __builtin_amdgcn_ds_swizzle(__builtin_bit_cast(int, x), 0x401F));
}

// Broadcast per-query-row scalar to the lanes that own that row in the C/D
// accumulator layout: acc vgpr v belongs to row (v + 8*half); the scalar for
// row i lives in lane i (and i+16). ds_bpermute: dst[lane] = src[idx[lane]].
static __device__ __forceinline__ float bcast_row(float x, int v) {
  int src = v + ((threadIdx.x & 16) >> 1);   // v + half*8, in 0..15
  return __builtin_bit_cast(float,
      __builtin_amdgcn_ds_bpermute(src << 2, __builtin_bit_cast(int, x)));
}

// ---------------------------------------------------------------------------
// f32 -> bf16 conversion, vectorized by 4.
// ---------------------------------------------------------------------------
__global__ void cvt_f32_bf16(const float* __restrict__ in,
                             bf16* __restrict__ out, int n4) {
  int i = blockIdx.x * blockDim.x + threadIdx.x;
  if (i < n4) {
    f32x4 v = reinterpret_cast<const f32x4*>(in)[i];
    bf16x4 o;
    o[0] = (bf16)v[0]; o[1] = (bf16)v[1]; o[2] = (bf16)v[2]; o[3] = (bf16)v[3];
    reinterpret_cast<bf16x4*>(out)[i] = o;
  }
}

// ---------------------------------------------------------------------------
// GEMM: Y[M=4096][N=1024] = X[M][K] * W[N][K]^T   (K = 1024, bf16 in, f32 acc)
// Block = 256 threads = 8 waves (2 M-waves x 4 N-waves), block tile 128x128,
// wave tile 64x32 = 4x2 WMMA tiles.
// MODE 0: bf16 out, row-major, *scale
// MODE 1: bf16 out, per-head transposed -> Vt[b][h][d][tok]
// MODE 2: f32 out, row-major, *scale
// ---------------------------------------------------------------------------
template <int MODE>
__global__ __launch_bounds__(256)
void gemm_wmma(const bf16* __restrict__ X, const bf16* __restrict__ W,
               void* __restrict__ Y, float scale) {
  constexpr int K = kDIM;
  const int lane = threadIdx.x & 31;
  const int wave = threadIdx.x >> 5;
  const int half = lane >> 4;            // 0|1
  const int l16  = lane & 15;
  const int m0 = blockIdx.y * 128 + (wave >> 2) * 64;
  const int n0 = blockIdx.x * 128 + (wave & 3) * 32;

  const bf16* xrow[4];
#pragma unroll
  for (int t = 0; t < 4; ++t) xrow[t] = X + (size_t)(m0 + t * 16 + l16) * K;
  const bf16* wrow[2];
#pragma unroll
  for (int t = 0; t < 2; ++t) wrow[t] = W + (size_t)(n0 + t * 16 + l16) * K;
  const int ka = half * 8;    // A-frag: k elems {ka..ka+7, ka+16..ka+23}
  const int kb = half * 8;    // B-frag: k elems {kb..kb+7, kb+16..kb+23}

  f32x8 acc[4][2] = {};
  for (int k0 = 0; k0 < K; k0 += 32) {
    bf16x16 a[4], b[2];
#pragma unroll
    for (int t = 0; t < 4; ++t) {
      bf16x8 lo = *reinterpret_cast<const bf16x8*>(xrow[t] + k0 + ka);
      bf16x8 hi = *reinterpret_cast<const bf16x8*>(xrow[t] + k0 + ka + 16);
      a[t] = join16(lo, hi);
    }
#pragma unroll
    for (int t = 0; t < 2; ++t) {
      bf16x8 lo = *reinterpret_cast<const bf16x8*>(wrow[t] + k0 + kb);
      bf16x8 hi = *reinterpret_cast<const bf16x8*>(wrow[t] + k0 + kb + 16);
      b[t] = join16(lo, hi);
    }
#pragma unroll
    for (int mt = 0; mt < 4; ++mt)
#pragma unroll
      for (int nt = 0; nt < 2; ++nt)
        acc[mt][nt] = wmma_bf16(a[mt], b[nt], acc[mt][nt]);
  }

#pragma unroll
  for (int mt = 0; mt < 4; ++mt)
#pragma unroll
    for (int nt = 0; nt < 2; ++nt)
#pragma unroll
      for (int v = 0; v < 8; ++v) {
        const int m = m0 + mt * 16 + v + half * 8;
        const int n = n0 + nt * 16 + l16;
        const float val = acc[mt][nt][v] * scale;
        if constexpr (MODE == 2) {
          reinterpret_cast<float*>(Y)[(size_t)m * kDIM + n] = val;
        } else if constexpr (MODE == 0) {
          reinterpret_cast<bf16*>(Y)[(size_t)m * kDIM + n] = (bf16)val;
        } else {  // Vt[b][h][d][tok]
          const int bb = m >> 11, tok = m & (kL - 1);
          const int h = n >> 6, d = n & (kHD - 1);
          reinterpret_cast<bf16*>(Y)[(((size_t)bb * kNH + h) * kHD + d) * kL + tok] =
              (bf16)val;
        }
      }
}

// ---------------------------------------------------------------------------
// Flash attention, exp2 domain (Q pre-scaled by 0.125*log2e).
// Grid: B*H*(L/128) blocks of 256 threads; each wave owns 16 query rows and
// streams keys in blocks of 32.  S^T = K Q^T so query index = lane%16; the
// C-layout of two S^T tiles IS the A-fragment layout of P(16x32).
// ---------------------------------------------------------------------------
__global__ __launch_bounds__(256)
void attn_wmma(const bf16* __restrict__ Q, const bf16* __restrict__ Km,
               const bf16* __restrict__ Vt, bf16* __restrict__ O) {
  const int lane = threadIdx.x & 31;
  const int wave = threadIdx.x >> 5;
  const int half = lane >> 4;
  const int l16  = lane & 15;
  const int blk  = blockIdx.x;
  const int hb   = blk >> 4;                       // b*16 + h
  const int i0   = (blk & 15) * 128 + wave * 16;   // query row block
  const int b    = hb >> 4, h = hb & 15;

  const size_t qkBase = (size_t)b * kL * kDIM + (size_t)h * kHD;
  const size_t vtBase = (size_t)hb * kHD * kL;

  // Q^T B-fragments (reused for every key block): lane col = query i0+l16.
  bf16x16 bq[2];
#pragma unroll
  for (int t = 0; t < 2; ++t) {
    const bf16* p = Q + qkBase + (size_t)(i0 + l16) * kDIM + t * 32 + half * 8;
    bq[t] = join16(*reinterpret_cast<const bf16x8*>(p),
                   *reinterpret_cast<const bf16x8*>(p + 16));
  }

  f32x8 acc[4] = {};
  float M = -1e30f, lsum = 0.f;

  for (int j0 = 0; j0 < kL; j0 += 32) {
    // S^T tiles (16 keys x 16 queries), contraction over d = 64 -> 2 WMMAs each.
    f32x8 st[2] = {};
#pragma unroll
    for (int jt = 0; jt < 2; ++jt)
#pragma unroll
      for (int t = 0; t < 2; ++t) {
        const bf16* p = Km + qkBase + (size_t)(j0 + jt * 16 + l16) * kDIM +
                        t * 32 + half * 8;
        bf16x16 ak = join16(*reinterpret_cast<const bf16x8*>(p),
                            *reinterpret_cast<const bf16x8*>(p + 16));
        st[jt] = wmma_bf16(ak, bq[t], st[jt]);
      }

    // Online softmax over this 32-key block (exp2 domain).
    float bm = st[0][0];
#pragma unroll
    for (int e = 1; e < 8; ++e) bm = fmaxf(bm, st[0][e]);
#pragma unroll
    for (int e = 0; e < 8; ++e) bm = fmaxf(bm, st[1][e]);
    bm = fmaxf(bm, swz_xor16(bm));         // both half-groups of a query agree
    const float Mnew = fmaxf(M, bm);
    const float r = __builtin_amdgcn_exp2f(M - Mnew);
    M = Mnew;

    float ps = 0.f;
    bf16x16 pa;                            // A-fragment of P (16 x 32)
#pragma unroll
    for (int e = 0; e < 8; ++e) {
      const float p0 = __builtin_amdgcn_exp2f(st[0][e] - Mnew);
      const float p1 = __builtin_amdgcn_exp2f(st[1][e] - Mnew);
      ps += p0 + p1;
      pa[e]     = (bf16)p0;
      pa[e + 8] = (bf16)p1;
    }
    lsum = lsum * r + ps;

    // Rescale running output by per-row r, then accumulate P*V.
    float rv[8];
#pragma unroll
    for (int v = 0; v < 8; ++v) rv[v] = bcast_row(r, v);
#pragma unroll
    for (int t = 0; t < 4; ++t)
#pragma unroll
      for (int v = 0; v < 8; ++v) acc[t][v] *= rv[v];

#pragma unroll
    for (int t = 0; t < 4; ++t) {
      const bf16* p = Vt + vtBase + (size_t)(t * 16 + l16) * kL + j0 + half * 8;
      bf16x16 bv = join16(*reinterpret_cast<const bf16x8*>(p),
                          *reinterpret_cast<const bf16x8*>(p + 16));
      acc[t] = wmma_bf16(pa, bv, acc[t]);
    }
  }

  // Normalize and store bf16 rows: row i = i0 + v + half*8, col = t*16 + l16.
  lsum += swz_xor16(lsum);
#pragma unroll
  for (int v = 0; v < 8; ++v) {
    const float inv = __builtin_amdgcn_rcpf(bcast_row(lsum, v));
    bf16* orow = O + qkBase + (size_t)(i0 + v + half * 8) * kDIM;
#pragma unroll
    for (int t = 0; t < 4; ++t)
      orow[t * 16 + l16] = (bf16)(acc[t][v] * inv);
  }
}

// ---------------------------------------------------------------------------
extern "C" void kernel_launch(void* const* d_in, const int* in_sizes, int n_in,
                              void* d_out, int out_size, void* d_ws, size_t ws_size,
                              hipStream_t stream) {
  const float* q  = (const float*)d_in[0];
  const float* Wq = (const float*)d_in[1];
  const float* Wk = (const float*)d_in[2];
  const float* Wv = (const float*)d_in[3];
  const float* Wo = (const float*)d_in[4];

  constexpr size_t SZ_X = (size_t)kTOK * kDIM * 2;  // 8 MB bf16 activation
  constexpr size_t SZ_W = (size_t)kDIM * kDIM * 2;  // 2 MB bf16 weight
  char* ws = (char*)d_ws;
  bf16* qb    = (bf16*)(ws);
  bf16* Wqb   = (bf16*)(ws + SZ_X);
  bf16* Wkb   = (bf16*)(ws + SZ_X + SZ_W);
  bf16* Wvb   = (bf16*)(ws + SZ_X + 2 * SZ_W);
  bf16* Wob   = (bf16*)(ws + SZ_X + 3 * SZ_W);
  bf16* Qb    = (bf16*)(ws + SZ_X + 4 * SZ_W);
  bf16* Kb    = (bf16*)(ws + 2 * SZ_X + 4 * SZ_W);
  bf16* Vtb   = (bf16*)(ws + 3 * SZ_X + 4 * SZ_W);
  bf16* attnb = (bf16*)(ws + 4 * SZ_X + 4 * SZ_W);  // total 48 MB

  const int thr = 256;
  const int nq4 = kTOK * kDIM / 4;
  const int nw4 = kDIM * kDIM / 4;
  cvt_f32_bf16<<<(nq4 + thr - 1) / thr, thr, 0, stream>>>(q,  qb,  nq4);
  cvt_f32_bf16<<<(nw4 + thr - 1) / thr, thr, 0, stream>>>(Wq, Wqb, nw4);
  cvt_f32_bf16<<<(nw4 + thr - 1) / thr, thr, 0, stream>>>(Wk, Wkb, nw4);
  cvt_f32_bf16<<<(nw4 + thr - 1) / thr, thr, 0, stream>>>(Wv, Wvb, nw4);
  cvt_f32_bf16<<<(nw4 + thr - 1) / thr, thr, 0, stream>>>(Wo, Wob, nw4);

  dim3 gg(kDIM / 128, kTOK / 128);  // (8, 32)
  const float qscale = 0.125f * 1.44269504088896340736f;  // 1/sqrt(64) * log2(e)
  gemm_wmma<0><<<gg, thr, 0, stream>>>(qb, Wqb, (void*)Qb,  qscale);
  gemm_wmma<0><<<gg, thr, 0, stream>>>(qb, Wkb, (void*)Kb,  1.0f);
  gemm_wmma<1><<<gg, thr, 0, stream>>>(qb, Wvb, (void*)Vtb, 1.0f);

  attn_wmma<<<kB * kNH * (kL / 128), thr, 0, stream>>>(Qb, Kb, Vtb, attnb);

  gemm_wmma<2><<<gg, thr, 0, stream>>>(attnb, Wob, d_out, 1.0f);
}
